// VectorQuantiser_72327249264798
// MI455X (gfx1250) — matrix operand order repaired and checked by reading, not compiled, not run on hardware
//
#include <hip/hip_runtime.h>

typedef __attribute__((ext_vector_type(16))) _Float16 v16h;
typedef __attribute__((ext_vector_type(8)))  float    v8f;

#define D_DIM     64
#define K_CODES   512
#define KT_TILES  32        // 512 / 16
#define OUT_ELEMS 4194304   // 64*32*32*64

// workspace layout (bytes)
#define WS_EMBT_OFF  0        // 512*64*4  = 131072
#define WS_EMBB_OFF  131072   // 32768*2   = 65536
#define WS_ENORM_OFF 196608   // 512*4     = 2048
#define WS_PART_OFF  198656   // 256*4     = 1024

// ---------------------------------------------------------------------------
// Prep 1: transpose codebook to [K][D] fp32 (coalesced gather) + ||e_k||^2
// ---------------------------------------------------------------------------
__global__ void vq_prep_embT(const float* __restrict__ emb,
                             float* __restrict__ embT,
                             float* __restrict__ enorm) {
  const int k = blockIdx.x;   // 0..511
  const int d = threadIdx.x;  // 0..63
  __shared__ float s[64];
  const float v = emb[d * K_CODES + k];
  embT[k * D_DIM + d] = v;
  s[d] = v * v;
  __syncthreads();
  if (d == 0) {
    float t = 0.f;
    #pragma unroll
    for (int i = 0; i < 64; ++i) t += s[i];
    enorm[k] = t;
  }
}

// ---------------------------------------------------------------------------
// Prep 2: convert codebook to f16, pre-swizzled into WMMA B-operand layout.
// 16-bit B (32x16), contiguous k order used here: lanes 0-15 hold Kdim=0..15 (elem e -> Kdim=e), lanes
// 16-31 hold Kdim=16..31; lane%16 = column (code) index within tile.
// flat = ((kt*2 + s)*32 + lane)*16 + e
// ---------------------------------------------------------------------------
__global__ void vq_prep_embB(const float* __restrict__ emb,
                             _Float16* __restrict__ embB) {
  const int t  = blockIdx.x * blockDim.x + threadIdx.x; // 0..32767
  const int e  = t & 15;
  const int l  = (t >> 4) & 31;
  const int s  = (t >> 9) & 1;
  const int kt = t >> 10;
  const int dl = ((l >> 4) << 4) + e;      // Kdim within 32
  const int d  = s * 32 + dl;              // reduction index (0..63)
  const int k  = kt * 16 + (l & 15);       // code index (0..511)
  embB[t] = (_Float16)emb[d * K_CODES + k];
}

// ---------------------------------------------------------------------------
// Main: 32 rows of x per wave (two 16-row M-tiles sharing each B load);
// 32 code tiles, 4 x v_wmma_f32_16x16x32_f16 per tile, B double-buffered.
// ---------------------------------------------------------------------------

#define BUILD_A(AREG0, AREG1, ROWBASE)                                        \
  do {                                                                        \
    const float* xrow_ = x + (size_t)((ROWBASE) + m) * D_DIM;                 \
    _Pragma("unroll")                                                         \
    for (int j = 0; j < 8; ++j) {                                             \
      const int dl_ = 2 * j + 16 * half;                                      \
      const float2 f0_ = *(const float2*)(xrow_ + dl_);                       \
      const float2 f1_ = *(const float2*)(xrow_ + 32 + dl_);                  \
      AREG0[2 * j]     = (_Float16)f0_.x;                                     \
      AREG0[2 * j + 1] = (_Float16)f0_.y;                                     \
      AREG1[2 * j]     = (_Float16)f1_.x;                                     \
      AREG1[2 * j + 1] = (_Float16)f1_.y;                                     \
    }                                                                         \
  } while (0)

#define EMIT(ROW, IDX)                                                        \
  do {                                                                        \
    const float2 q_  = *(const float2*)(embT + (IDX) * D_DIM + d0);           \
    const float2 xv_ = *(const float2*)(x + (size_t)(ROW) * D_DIM + d0);      \
    const float dx_ = q_.x - xv_.x, dy_ = q_.y - xv_.y;                       \
    float2 o_; o_.x = xv_.x + dx_; o_.y = xv_.y + dy_;                        \
    *(float2*)(out + (size_t)(ROW) * D_DIM + d0) = o_;                        \
    lsum += dx_ * dx_ + dy_ * dy_;                                            \
  } while (0)

__global__ void __launch_bounds__(256)
vq_main(const float* __restrict__ x,
        const _Float16* __restrict__ embB,
        const float* __restrict__ embT,
        const float* __restrict__ enorm,
        float* __restrict__ out,
        float* __restrict__ partials) {
  const int lane = threadIdx.x & 31;
  const int wave = threadIdx.x >> 5;
  const int row0 = (blockIdx.x * 8 + wave) * 32;   // first M-tile
  const int row1 = row0 + 16;                      // second M-tile

  const int m    = lane & 15;
  const int half = lane >> 4;

  // A operands for both 16-row tiles (16-bit A 16x32, contiguous k = 16*half + e, same as B)
  v16h a0, a1, a2, a3;
  BUILD_A(a0, a1, row0);
  BUILD_A(a2, a3, row1);

  float minv0[8], minv1[8];
  int   mini0[8], mini1[8];
  #pragma unroll
  for (int r = 0; r < 8; ++r) {
    minv0[r] = 3.4e38f; mini0[r] = 0;
    minv1[r] = 3.4e38f; mini1[r] = 0;
  }

  // software pipeline: preload tile 0's B operands and ||e||^2
  v16h  b0 = *(const v16h*)(embB + (0 * 32 + lane) * 16);
  v16h  b1 = *(const v16h*)(embB + (1 * 32 + lane) * 16);
  float en = enorm[m];

  #pragma unroll 2
  for (int kt = 0; kt < KT_TILES; ++kt) {
    const int ktn = (kt + 1) & (KT_TILES - 1);   // wraps: last prefetch unused
    const v16h  nb0 = *(const v16h*)(embB + ((ktn * 2 + 0) * 32 + lane) * 16);
    const v16h  nb1 = *(const v16h*)(embB + ((ktn * 2 + 1) * 32 + lane) * 16);
    const float nen = enorm[ktn * 16 + m];

    v8f c0 = {};
    c0 = __builtin_amdgcn_wmma_f32_16x16x32_f16(false, a0, false, b0,
                                                (short)0, c0, false, false);
    c0 = __builtin_amdgcn_wmma_f32_16x16x32_f16(false, a1, false, b1,
                                                (short)0, c0, false, false);
    v8f c1 = {};
    c1 = __builtin_amdgcn_wmma_f32_16x16x32_f16(false, a2, false, b0,
                                                (short)0, c1, false, false);
    c1 = __builtin_amdgcn_wmma_f32_16x16x32_f16(false, a3, false, b1,
                                                (short)0, c1, false, false);

    // C layout: lane holds column n = lane&15; rows (lane>>4)*8 + r
    const int kcol = kt * 16 + m;
    #pragma unroll
    for (int r = 0; r < 8; ++r) {
      const float d0_ = en - 2.0f * c0[r];   // ||e||^2 - 2 x.e (argmin-equiv)
      if (d0_ < minv0[r]) { minv0[r] = d0_; mini0[r] = kcol; }
      const float d1_ = en - 2.0f * c1[r];
      if (d1_ < minv1[r]) { minv1[r] = d1_; mini1[r] = kcol; }
    }

    b0 = nb0; b1 = nb1; en = nen;
  }

  // argmin across the 16 column-lanes of each half-group (lowest-index ties)
  #pragma unroll
  for (int off = 1; off < 16; off <<= 1) {
    #pragma unroll
    for (int r = 0; r < 8; ++r) {
      {
        const float ov = __shfl_xor(minv0[r], off, 32);
        const int   oi = __shfl_xor(mini0[r], off, 32);
        if (ov < minv0[r] || (ov == minv0[r] && oi < mini0[r])) {
          minv0[r] = ov; mini0[r] = oi;
        }
      }
      {
        const float ov = __shfl_xor(minv1[r], off, 32);
        const int   oi = __shfl_xor(mini1[r], off, 32);
        if (ov < minv1[r] || (ov == minv1[r] && oi < mini1[r])) {
          minv1[r] = ov; mini1[r] = oi;
        }
      }
    }
  }

  // Gather codebook rows, write out = x + (q - x), accumulate (q - x)^2
  float lsum = 0.f;
  const int d0 = lane * 2;
  #pragma unroll
  for (int r = 0; r < 8; ++r) {
    {
      const int iA = __shfl(mini0[r], 0, 32);   // row row0 + r
      const int iB = __shfl(mini0[r], 16, 32);  // row row0 + 8 + r
      EMIT(row0 + r, iA);
      EMIT(row0 + 8 + r, iB);
    }
    {
      const int iA = __shfl(mini1[r], 0, 32);   // row row1 + r
      const int iB = __shfl(mini1[r], 16, 32);  // row row1 + 8 + r
      EMIT(row1 + r, iA);
      EMIT(row1 + 8 + r, iB);
    }
  }

  // deterministic loss reduction: wave -> block -> per-block partial
  #pragma unroll
  for (int off = 16; off >= 1; off >>= 1) lsum += __shfl_xor(lsum, off, 32);

  __shared__ float wsum[8];
  if (lane == 0) wsum[wave] = lsum;
  __syncthreads();
  if (threadIdx.x == 0) {
    float t = 0.f;
    #pragma unroll
    for (int i = 0; i < 8; ++i) t += wsum[i];
    partials[blockIdx.x] = t;
  }
}

// ---------------------------------------------------------------------------
// Finalize: sum 256 block partials -> loss = 2 * sum / (N*D)
// ---------------------------------------------------------------------------
__global__ void vq_finalize(const float* __restrict__ partials,
                            float* __restrict__ loss_out) {
  __shared__ float s[256];
  const int t = threadIdx.x;
  s[t] = partials[t];
  __syncthreads();
  for (int off = 128; off > 0; off >>= 1) {
    if (t < off) s[t] += s[t + off];
    __syncthreads();
  }
  if (t == 0) loss_out[0] = 2.0f * s[0] / (float)OUT_ELEMS;
}

// ---------------------------------------------------------------------------
extern "C" void kernel_launch(void* const* d_in, const int* in_sizes, int n_in,
                              void* d_out, int out_size, void* d_ws, size_t ws_size,
                              hipStream_t stream) {
  const float* x   = (const float*)d_in[0];   // [64,32,32,64] f32
  const float* emb = (const float*)d_in[1];   // [64,512]      f32
  float* out = (float*)d_out;                 // out (4194304) then loss (1)
  char*  ws  = (char*)d_ws;

  float*    embT  = (float*)   (ws + WS_EMBT_OFF);
  _Float16* embB  = (_Float16*)(ws + WS_EMBB_OFF);
  float*    enorm = (float*)   (ws + WS_ENORM_OFF);
  float*    parts = (float*)   (ws + WS_PART_OFF);

  vq_prep_embT<<<512, 64, 0, stream>>>(emb, embT, enorm);
  vq_prep_embB<<<128, 256, 0, stream>>>(emb, embB);
  vq_main<<<256, 256, 0, stream>>>(x, embB, embT, enorm, out, parts);
  vq_finalize<<<1, 256, 0, stream>>>(parts, out + OUT_ELEMS);
}
